// TransformerBlock_11579231830231
// MI455X (gfx1250) — matrix operand order repaired and checked by reading, not compiled, not run on hardware
//
#include <hip/hip_runtime.h>
#include <math.h>

// ---------------- types / constants ----------------
typedef unsigned short ushort_t;
typedef unsigned int   uint32;
typedef __attribute__((ext_vector_type(16))) __bf16 v16bf;
typedef __attribute__((ext_vector_type(8)))  float  v8f;
typedef int v4i __attribute__((vector_size(16)));   // matches builtin param type

#define T_TOK   1024
#define H_DIM   2880
#define Q_SIZE  4096
#define KV_SIZE 512
#define QKV_N   5120
#define NQH     64
#define NKVH    8
#define HEADD   64
#define WIN     128
#define NEXP    16
#define I_DIM   2880

static __device__ __forceinline__ float bf2f(ushort_t h) {
  return __uint_as_float(((uint32)h) << 16);
}
static __device__ __forceinline__ ushort_t f2bf(float f) {
  uint32 u = __float_as_uint(f);
  u += 0x7fffu + ((u >> 16) & 1u);           // round-to-nearest-even
  return (ushort_t)(u >> 16);
}

union Frag { v16bf v; uint32 u[8]; uint4 q[2]; };

// CDNA5 async global->LDS staging (guarded; falls back to load+ds_store)
#if defined(__has_builtin)
#if __has_builtin(__builtin_amdgcn_global_load_async_to_lds_b128) && \
    __has_builtin(__builtin_amdgcn_s_wait_asynccnt)
#define USE_ASYNC_LDS 1
#endif
#endif

#ifdef USE_ASYNC_LDS
// signature (confirmed via hipcc): (v4i AS1*, v4i AS3*, imm int offset, imm int cpol)
#define GLD_ASYNC_B128(gsrc, ldst)                                   \
  __builtin_amdgcn_global_load_async_to_lds_b128(                    \
      (__attribute__((address_space(1))) v4i*)(void*)(gsrc),         \
      (__attribute__((address_space(3))) v4i*)(void*)(ldst), 0, 0)
#endif

// ---------------- RMSNorm ----------------
__global__ __launch_bounds__(256)
void rmsnorm_kernel(const ushort_t* __restrict__ x, const ushort_t* __restrict__ w,
                    ushort_t* __restrict__ out, int H) {
  const int row = blockIdx.x;
  const int tid = threadIdx.x;
  const ushort_t* xr = x + (size_t)row * H;
  float ss = 0.f;
  for (int i = tid; i < H; i += 256) { float v = bf2f(xr[i]); ss += v * v; }
  __shared__ float red[256];
  red[tid] = ss; __syncthreads();
  for (int s = 128; s > 0; s >>= 1) { if (tid < s) red[tid] += red[tid + s]; __syncthreads(); }
  const float inv = rsqrtf(red[0] / (float)H + 1e-5f);
  ushort_t* orow = out + (size_t)row * H;
  for (int i = tid; i < H; i += 256) {
    float y = bf2f(xr[i]) * inv;
    orow[i] = f2bf(bf2f(f2bf(y)) * bf2f(w[i]));   // match bf16 cast then * w
  }
}

// ---------------- WMMA GEMM: C[M,N] = A[M,K] @ B[K,N] + bias ----------------
// Block tile 128(M) x 64(N), BK=32. 8 waves in 4x2; each wave owns 32x32
// (2x2 WMMA 16x16 tiles -> 4 v_wmma per K-step with A/B fragment reuse).
// A tile staged via global_load_async_to_lds_b128 (CDNA5 async DMA).
// B tile transposed in registers (pairs of K-rows) -> full-width ds_store_b32.
// mode 0: out bf16 = AB+bias
// mode 1: out bf16 = AB+bias+residual
// mode 2: out f32 += gates[row,expert] * (AB+bias)
__global__ __launch_bounds__(256)
void gemm_bf16_wmma(const ushort_t* __restrict__ A, const ushort_t* __restrict__ B,
                    const ushort_t* __restrict__ bias, void* __restrict__ Cout,
                    const ushort_t* __restrict__ residual, const float* __restrict__ gates,
                    int M, int N, int K, int mode, int expert) {
  __shared__ ushort_t As[128 * 32];  // [m][k], stride 32 (8 KB)
  __shared__ ushort_t BsT[64 * 32];  // [n][k], stride 32, (k,k+1) pairs per dword (4 KB)

  const int tid  = threadIdx.x;
  const int lane = tid & 31, wave = tid >> 5;
  const int wr = wave >> 1;               // wave M strip (0..3) of 32 rows
  const int wc = wave & 1;                // wave N strip (0..1) of 32 cols
  const int l15 = lane & 15, lh = lane >> 4;
  const int bm0 = blockIdx.y * 128, bn0 = blockIdx.x * 64;

  v8f acc[2][2] = {};

  // staging assignments
  const int ar = tid >> 1, ac = (tid & 1) << 4;   // A: 128 rows, 16 elems/thread
  const int qr = tid >> 4, qc = (tid & 15) << 2;  // B: k-pair row qr (k=2qr,2qr+1), 4 cols

  const ushort_t* Aptr  = A + (size_t)(bm0 + ar) * K + ac;       // +32 per iter
  const ushort_t* Bptr0 = B + (size_t)(2 * qr) * N + (bn0 + qc); // +32*N per iter
  const ushort_t* Bptr1 = Bptr0 + N;
  ushort_t* AsDst = &As[ar * 32 + ac];
  uint32* Bst = (uint32*)BsT;

  for (int k0 = 0; k0 < K; k0 += 32) {
    // ---- stage A tile via async global->LDS DMA (no transpose needed) ----
#ifdef USE_ASYNC_LDS
    GLD_ASYNC_B128(Aptr, AsDst);
    GLD_ASYNC_B128(Aptr + 8, AsDst + 8);
#else
    {
      uint4 av0 = *(const uint4*)(Aptr);
      uint4 av1 = *(const uint4*)(Aptr + 8);
      *(uint4*)(AsDst)     = av0;
      *(uint4*)(AsDst + 8) = av1;
    }
#endif
    // ---- stage B tile: transpose k-row pairs in registers, b32 LDS stores ----
    {
      uint2 blo = *(const uint2*)(Bptr0);   // row k=2qr,  cols qc..qc+3
      uint2 bhi = *(const uint2*)(Bptr1);   // row k=2qr+1, cols qc..qc+3
      Bst[(qc + 0) * 16 + qr] = (blo.x & 0xffffu) | (bhi.x << 16);
      Bst[(qc + 1) * 16 + qr] = (blo.x >> 16)     | (bhi.x & 0xffff0000u);
      Bst[(qc + 2) * 16 + qr] = (blo.y & 0xffffu) | (bhi.y << 16);
      Bst[(qc + 3) * 16 + qr] = (blo.y >> 16)     | (bhi.y & 0xffff0000u);
    }
    // prefetch next K tiles into cache
    if (k0 + 32 < K) {
      __builtin_prefetch(Aptr + 32, 0, 1);
      __builtin_prefetch(Bptr0 + (size_t)32 * N, 0, 1);
    }
    Aptr  += 32;
    Bptr0 += (size_t)32 * N;
    Bptr1 += (size_t)32 * N;
#ifdef USE_ASYNC_LDS
    __builtin_amdgcn_s_wait_asynccnt(0);
#endif
    __syncthreads();

    // ---- fragments (CDNA5 16-bit WMMA layouts), b128 LDS reads ----
    const uint32* As32 = (const uint32*)As;
    const uint32* Bs32 = (const uint32*)BsT;
    Frag fa[2], fb[2];
#pragma unroll
    for (int tm = 0; tm < 2; ++tm) {
      const int am = wr * 32 + tm * 16 + l15;     // row within block tile
      // lanes 0-15: K 0..7 / 16..23 ; lanes 16-31: K 8..15 / 24..31
      fa[tm].q[0] = *(const uint4*)(As32 + am * 16 + lh * 4);
      fa[tm].q[1] = *(const uint4*)(As32 + am * 16 + 8 + lh * 4);
    }
#pragma unroll
    for (int tn = 0; tn < 2; ++tn) {
      const int bn = wc * 32 + tn * 16 + l15;     // col within block tile
      // lanes 0-15: K 0..7 / 16..23 ; lanes 16-31: K 8..15 / 24..31
      fb[tn].q[0] = *(const uint4*)(Bs32 + bn * 16 + lh * 4);
      fb[tn].q[1] = *(const uint4*)(Bs32 + bn * 16 + 8 + lh * 4);
    }
#pragma unroll
    for (int tm = 0; tm < 2; ++tm)
#pragma unroll
      for (int tn = 0; tn < 2; ++tn)
        acc[tm][tn] = __builtin_amdgcn_wmma_f32_16x16x32_bf16(
            false, fa[tm].v, false, fb[tn].v, (short)0, acc[tm][tn], false, false);
    __syncthreads();
  }

  // epilogue: C/D layout — lane = N, VGPR i -> M = i + (lane>=16 ? 8 : 0)
#pragma unroll
  for (int tm = 0; tm < 2; ++tm) {
#pragma unroll
    for (int tn = 0; tn < 2; ++tn) {
      const int col = bn0 + wc * 32 + tn * 16 + l15;
      const float bz = bf2f(bias[col]);
#pragma unroll
      for (int i = 0; i < 8; ++i) {
        const int row = bm0 + wr * 32 + tm * 16 + lh * 8 + i;
        float v = acc[tm][tn][i] + bz;
        const size_t off = (size_t)row * N + col;
        if (mode == 2) {
          ((float*)Cout)[off] += gates[(size_t)row * NEXP + expert] * v;
        } else {
          if (mode == 1) v += bf2f(residual[off]);
          ((ushort_t*)Cout)[off] = f2bf(v);
        }
      }
    }
  }
}

// ---------------- YaRN RoPE (NeoX) on q and k inside qkv ----------------
__global__ __launch_bounds__(32)
void rope_kernel(ushort_t* __restrict__ qkv, const int* __restrict__ positions) {
  const int t = blockIdx.x;
  const int h = blockIdx.y;            // 0..63 q heads, 64..71 k heads
  const int i = threadIdx.x;           // 0..31 frequency index
  ushort_t* base = (h < NQH)
      ? qkv + (size_t)t * QKV_N + h * HEADD
      : qkv + (size_t)t * QKV_N + Q_SIZE + (h - NQH) * HEADD;

  const float ex = (2.0f * (float)i) / (float)HEADD;
  const float pf = powf(150000.0f, ex);
  const float inv_extra = 1.0f / pf;
  const float inv_inter = 1.0f / (32.0f * pf);
  const float two_pi = 6.2831853071795864769f;
  const float denomc = 2.0f * logf(150000.0f);
  float low  = floorf((float)HEADD * logf(4096.0f / (32.0f * two_pi)) / denomc);
  float high = ceilf ((float)HEADD * logf(4096.0f / ( 1.0f * two_pi)) / denomc);
  low  = fmaxf(low, 0.0f);
  high = fminf(high, (float)HEADD - 1.0f);
  float ramp = ((float)i - low) / fmaxf(high - low, 0.001f);
  ramp = fminf(fmaxf(ramp, 0.0f), 1.0f);
  const float mask = 1.0f - ramp;
  const float invf = inv_inter * (1.0f - mask) + inv_extra * mask;
  const float mscale = 0.1f * logf(32.0f) + 1.0f;
  const float fr = (float)positions[t] * invf;
  const float c = cosf(fr) * mscale, s = sinf(fr) * mscale;
  const float x1 = bf2f(base[i]), x2 = bf2f(base[i + 32]);
  base[i]      = f2bf(x1 * c - x2 * s);
  base[i + 32] = f2bf(x2 * c + x1 * s);
}

// ---------------- sliding-window attention with sinks ----------------
__global__ __launch_bounds__(128)
void attn_kernel(const ushort_t* __restrict__ qkv, const ushort_t* __restrict__ sinks,
                 ushort_t* __restrict__ obuf) {
  const int t = blockIdx.x;
  const int h = blockIdx.y;
  const int kvh = h >> 3;
  const int tid = threadIdx.x;     // 0..127, one key per thread
  __shared__ float qs[HEADD];
  __shared__ float ps[WIN];
  __shared__ float sred[WIN];

  const ushort_t* qp = qkv + (size_t)t * QKV_N + h * HEADD;
  if (tid < HEADD) qs[tid] = bf2f(qp[tid]);
  __syncthreads();

  const int nk = (t + 1 < WIN) ? (t + 1) : WIN;
  float score = -1e30f;
  if (tid < nk) {
    const int j = t - tid;
    const ushort_t* kp = qkv + (size_t)j * QKV_N + Q_SIZE + kvh * HEADD;
    float s = 0.f;
#pragma unroll 8
    for (int d = 0; d < HEADD; ++d) s += qs[d] * bf2f(kp[d]);
    score = s * 0.125f;            // 1/sqrt(64)
  }
  sred[tid] = score; __syncthreads();
  for (int s = 64; s > 0; s >>= 1) { if (tid < s) sred[tid] = fmaxf(sred[tid], sred[tid + s]); __syncthreads(); }
  const float snk = bf2f(sinks[h]);
  const float m = fmaxf(sred[0], snk);
  __syncthreads();
  const float p = (tid < nk) ? expf(score - m) : 0.f;
  ps[tid] = p; sred[tid] = p; __syncthreads();
  for (int s = 64; s > 0; s >>= 1) { if (tid < s) sred[tid] += sred[tid + s]; __syncthreads(); }
  const float denom = sred[0] + expf(snk - m);

  if (tid < HEADD) {
    float o = 0.f;
    for (int kk = 0; kk < nk; ++kk) {
      const int j = t - kk;
      o += ps[kk] * bf2f(qkv[(size_t)j * QKV_N + Q_SIZE + KV_SIZE + kvh * HEADD + tid]);
    }
    obuf[(size_t)t * Q_SIZE + h * HEADD + tid] = f2bf(o / denom);
  }
}

// ---------------- router: softmax + top-4 renormalized gates ----------------
__global__ __launch_bounds__(32)
void router_kernel(const ushort_t* __restrict__ t2, const ushort_t* __restrict__ rw,
                   const ushort_t* __restrict__ rb, float* __restrict__ gates) {
  const int t = blockIdx.x;
  const int e = threadIdx.x;
  __shared__ float g[NEXP];
  if (e < NEXP) {
    float s = bf2f(rb[e]);
    const ushort_t* xr = t2 + (size_t)t * H_DIM;
    for (int i = 0; i < H_DIM; ++i) s += bf2f(xr[i]) * bf2f(rw[i * NEXP + e]);
    g[e] = s;
  }
  __syncthreads();
  if (e == 0) {
    float m = -1e30f;
    for (int i = 0; i < NEXP; ++i) m = fmaxf(m, g[i]);
    float p[NEXP]; float sum = 0.f;
    for (int i = 0; i < NEXP; ++i) { p[i] = expf(g[i] - m); sum += p[i]; }
    for (int i = 0; i < NEXP; ++i) p[i] /= sum;
    float gate[NEXP]; bool used[NEXP];
    for (int i = 0; i < NEXP; ++i) { gate[i] = 0.f; used[i] = false; }
    float tsum = 0.f; int idx[4]; float val[4];
    for (int k = 0; k < 4; ++k) {
      int bi = -1; float bvv = -1.f;
      for (int i = 0; i < NEXP; ++i)
        if (!used[i] && p[i] > bvv) { bvv = p[i]; bi = i; }
      used[bi] = true; idx[k] = bi; val[k] = bvv; tsum += bvv;
    }
    for (int k = 0; k < 4; ++k) gate[idx[k]] = val[k] / tsum;
    for (int i = 0; i < NEXP; ++i) gates[(size_t)t * NEXP + i] = gate[i];
  }
}

// ---------------- SwiGLU with clamps ----------------
__global__ __launch_bounds__(256)
void swiglu_kernel(const ushort_t* __restrict__ h, ushort_t* __restrict__ act, int n) {
  const int idx = blockIdx.x * 256 + threadIdx.x;
  if (idx >= n) return;
  const int t = idx / I_DIM, i = idx - t * I_DIM;
  float gf = bf2f(h[(size_t)t * (2 * I_DIM) + 2 * i]);
  float uf = bf2f(h[(size_t)t * (2 * I_DIM) + 2 * i + 1]);
  gf = fminf(gf, 7.0f);
  uf = fminf(fmaxf(uf, -7.0f), 7.0f);
  const float sg = 1.0f / (1.0f + expf(-gf * 1.702f));
  act[(size_t)t * I_DIM + i] = f2bf((uf + 1.0f) * gf * sg);
}

// ---------------- misc ----------------
__global__ __launch_bounds__(256)
void zero_f32_kernel(float* __restrict__ p, int n) {
  const int i = blockIdx.x * 256 + threadIdx.x;
  if (i < n) p[i] = 0.f;
}
__global__ __launch_bounds__(256)
void final_add_kernel(const ushort_t* __restrict__ a, const float* __restrict__ macc,
                      ushort_t* __restrict__ out, int n) {
  const int i = blockIdx.x * 256 + threadIdx.x;
  if (i < n) out[i] = f2bf(bf2f(a[i]) + bf2f(f2bf(macc[i])));
}

// ---------------- host orchestration ----------------
extern "C" void kernel_launch(void* const* d_in, const int* in_sizes, int n_in,
                              void* d_out, int out_size, void* d_ws, size_t ws_size,
                              hipStream_t stream) {
  (void)in_sizes; (void)n_in; (void)out_size; (void)ws_size;
  const ushort_t* hidden   = (const ushort_t*)d_in[0];
  const int*      positions= (const int*)d_in[1];
  const ushort_t* norm1_w  = (const ushort_t*)d_in[2];
  const ushort_t* qkv_w    = (const ushort_t*)d_in[3];
  const ushort_t* qkv_b    = (const ushort_t*)d_in[4];
  const ushort_t* sinks    = (const ushort_t*)d_in[5];
  const ushort_t* o_w      = (const ushort_t*)d_in[6];
  const ushort_t* o_b      = (const ushort_t*)d_in[7];
  const ushort_t* norm2_w  = (const ushort_t*)d_in[8];
  const ushort_t* router_w = (const ushort_t*)d_in[9];
  const ushort_t* router_b = (const ushort_t*)d_in[10];
  const ushort_t* w13      = (const ushort_t*)d_in[11];
  const ushort_t* b13      = (const ushort_t*)d_in[12];
  const ushort_t* w2       = (const ushort_t*)d_in[13];
  const ushort_t* b2       = (const ushort_t*)d_in[14];

  char* ws = (char*)d_ws;
  size_t off = 0;
  auto alloc = [&](size_t bytes) -> void* {
    void* p = ws + off; off += (bytes + 255) & ~(size_t)255; return p;
  };
  ushort_t* t1   = (ushort_t*)alloc((size_t)T_TOK * H_DIM * 2);
  ushort_t* qkv  = (ushort_t*)alloc((size_t)T_TOK * QKV_N * 2);
  ushort_t* obuf = (ushort_t*)alloc((size_t)T_TOK * Q_SIZE * 2);
  ushort_t* abuf = (ushort_t*)alloc((size_t)T_TOK * H_DIM * 2);
  ushort_t* t2   = (ushort_t*)alloc((size_t)T_TOK * H_DIM * 2);
  float*    gts  = (float*)   alloc((size_t)T_TOK * NEXP * 4);
  ushort_t* hbuf = (ushort_t*)alloc((size_t)T_TOK * 2 * I_DIM * 2);
  ushort_t* actb = (ushort_t*)alloc((size_t)T_TOK * I_DIM * 2);
  float*    macc = (float*)   alloc((size_t)T_TOK * H_DIM * 4);

  // ---- attention block ----
  rmsnorm_kernel<<<T_TOK, 256, 0, stream>>>(hidden, norm1_w, t1, H_DIM);
  gemm_bf16_wmma<<<dim3(QKV_N / 64, T_TOK / 128), 256, 0, stream>>>(
      t1, qkv_w, qkv_b, qkv, nullptr, nullptr, T_TOK, QKV_N, H_DIM, 0, 0);
  rope_kernel<<<dim3(T_TOK, NQH + NKVH), 32, 0, stream>>>(qkv, positions);
  attn_kernel<<<dim3(T_TOK, NQH), 128, 0, stream>>>(qkv, sinks, obuf);
  gemm_bf16_wmma<<<dim3(H_DIM / 64, T_TOK / 128), 256, 0, stream>>>(
      obuf, o_w, o_b, abuf, hidden, nullptr, T_TOK, H_DIM, Q_SIZE, 1, 0);

  // ---- MoE block ----
  rmsnorm_kernel<<<T_TOK, 256, 0, stream>>>(abuf, norm2_w, t2, H_DIM);
  router_kernel<<<T_TOK, 32, 0, stream>>>(t2, router_w, router_b, gts);
  zero_f32_kernel<<<(T_TOK * H_DIM + 255) / 256, 256, 0, stream>>>(macc, T_TOK * H_DIM);
  for (int e = 0; e < NEXP; ++e) {
    gemm_bf16_wmma<<<dim3(2 * I_DIM / 64, T_TOK / 128), 256, 0, stream>>>(
        t2, w13 + (size_t)e * H_DIM * 2 * I_DIM, b13 + (size_t)e * 2 * I_DIM,
        hbuf, nullptr, nullptr, T_TOK, 2 * I_DIM, H_DIM, 0, 0);
    swiglu_kernel<<<(T_TOK * I_DIM + 255) / 256, 256, 0, stream>>>(
        hbuf, actb, T_TOK * I_DIM);
    gemm_bf16_wmma<<<dim3(H_DIM / 64, T_TOK / 128), 256, 0, stream>>>(
        actb, w2 + (size_t)e * I_DIM * H_DIM, b2 + (size_t)e * H_DIM,
        macc, nullptr, gts, T_TOK, H_DIM, I_DIM, 2, e);
  }
  final_add_kernel<<<(T_TOK * H_DIM + 255) / 256, 256, 0, stream>>>(
      abuf, macc, (ushort_t*)d_out, T_TOK * H_DIM);
}
